// PLEMultiTaskModel_89945205113461
// MI455X (gfx1250) — compile-verified, matrix operand order checked by reading, not run on hardware
//
#include <hip/hip_runtime.h>

typedef __attribute__((ext_vector_type(16))) __bf16 v16bf;
typedef __attribute__((ext_vector_type(8)))  __bf16 v8bf;
typedef __attribute__((ext_vector_type(8)))  float  v8f;

#define B_ROWS 65536

// bf16-element offsets inside the packed weight blob (d_ws)
#define OFF_W1   0        // 6 experts x (16 kt * 4 nt * 512) = 6*32768
#define OFF_W2   196608   // 6 x (2*2*512) = 6*2048
#define OFF_W3   208896   // 6 x (1*2*512) = 6*1024
#define OFF_G    215040   // 16*1*512 = 8192   (g1 cols 0-3, g2 cols 4-7, rest 0)
#define OFF_TW1A 223232   // 1*4*512 = 2048
#define OFF_TW2A 225280   // 1*4*512 = 2048
#define OFF_TW1B 227328   // 2*1*512 = 1024
#define OFF_TW2B 228352   // 2*1*512 = 1024
// total 229376 bf16 = 458752 bytes of workspace

__device__ __forceinline__ v8f wmma_bf16(v16bf a, v16bf b, v8f c) {
  return __builtin_amdgcn_wmma_f32_16x16x32_bf16(false, a, false, b, (short)0, c,
                                                 false, false);
}

// K index held by (lane, element j) of a 16-bit WMMA A/B fragment (16x32 / 32x16)
__device__ __forceinline__ int kmap(int lane, int j) {
  int v = j >> 1, h = j & 1;
  return ((lane >> 4) * 8) + ((v >> 2) * 16) + ((v & 3) * 2) + h;
}

// Pack a KxN row-major f32 matrix into WMMA B fragments:
// blob layout: [kt][nt][lane 0..31][16 bf16]  (32B contiguous per lane chunk)
__global__ void pack_b(const float* __restrict__ src, __bf16* __restrict__ dst,
                       int K, int N, int NT) {
  int total = (K >> 5) * NT * 512;
  int p = blockIdx.x * blockDim.x + threadIdx.x;
  if (p >= total) return;
  int j = p & 15;
  int lane = (p >> 4) & 31;
  int tile = p >> 9;
  int nt = tile % NT, kt = tile / NT;
  int kk = kt * 32 + kmap(lane, j);
  int nn = nt * 16 + (lane & 15);
  float val = (nn < N) ? src[kk * N + nn] : 0.f;
  dst[p] = (__bf16)val;
}

// Pack both 512x4 gate matrices into one 512x16 B blob (cols 0-3 g1, 4-7 g2)
__global__ void pack_gate(const float* __restrict__ g1W,
                          const float* __restrict__ g2W,
                          __bf16* __restrict__ dst) {
  int p = blockIdx.x * blockDim.x + threadIdx.x;   // 8192 total
  if (p >= 8192) return;
  int j = p & 15, lane = (p >> 4) & 31, kt = p >> 9;
  int kk = kt * 32 + kmap(lane, j);
  int nn = lane & 15;
  float val = (nn < 4) ? g1W[kk * 4 + nn]
            : (nn < 8) ? g2W[kk * 4 + (nn - 4)] : 0.f;
  dst[p] = (__bf16)val;
}

struct PLEParams {
  const float* x;
  const __bf16* blob;
  const float *sh_b1, *t1_b1, *t2_b1;
  const float *sh_b2, *t1_b2, *t2_b2;
  const float *sh_b3, *t1_b3, *t2_b3;
  const float *g1b, *g2b;
  const float *tw1b1, *tw1b2, *tw2b1, *tw2b2;
  float* out;
};

__device__ __forceinline__ v16bf lds_afrag(const __bf16* rowp, int off) {
  union { v16bf v; v8bf h[2]; } u;
  u.h[0] = *(const v8bf*)(rowp + off);
  u.h[1] = *(const v8bf*)(rowp + off + 16);
  return u.v;
}

__device__ __forceinline__ v16bf bfrag(const __bf16* mat, int tile, int lane) {
  return *(const v16bf*)(mat + ((tile * 32 + lane) << 4));
}

#define WAVES 8
#define LDH1  72   // 64 cols + 8 pad (conflict-free b128 A reads)
#define LDH2  40   // 32 cols + 8 pad

__global__ __launch_bounds__(256) void ple_fused(PLEParams P) {
  __shared__ __align__(16) __bf16 hb [WAVES * 16 * LDH1];
  __shared__ __align__(16) __bf16 h2b[WAVES * 16 * LDH2];

  const int tid   = threadIdx.x;
  const int lane  = tid & 31;
  const int wave  = tid >> 5;
  const int half  = lane >> 4;
  const int col   = lane & 15;
  const int khalf = half * 8;
  const int rowBase = blockIdx.x * (WAVES * 16) + wave * 16;
  __bf16* hw  = hb  + wave * 16 * LDH1;
  __bf16* h2w = h2b + wave * 16 * LDH2;

  const v8f zacc = {0.f, 0.f, 0.f, 0.f, 0.f, 0.f, 0.f, 0.f};

  // ---- Preload all 16 K-tiles of this wave's 16x512 A matrix (f32 -> bf16) ----
  v16bf ax[16];
  const float* xr = P.x + (size_t)(rowBase + col) * 512 + khalf;
  #pragma unroll
  for (int kt = 0; kt < 16; ++kt) {
    const float4* q = (const float4*)(xr + kt * 32);
    float4 f0 = q[0], f1 = q[1], f2 = q[4], f3 = q[5];
    v16bf a;
    a[0]=(__bf16)f0.x;  a[1]=(__bf16)f0.y;  a[2]=(__bf16)f0.z;  a[3]=(__bf16)f0.w;
    a[4]=(__bf16)f1.x;  a[5]=(__bf16)f1.y;  a[6]=(__bf16)f1.z;  a[7]=(__bf16)f1.w;
    a[8]=(__bf16)f2.x;  a[9]=(__bf16)f2.y;  a[10]=(__bf16)f2.z; a[11]=(__bf16)f2.w;
    a[12]=(__bf16)f3.x; a[13]=(__bf16)f3.y; a[14]=(__bf16)f3.z; a[15]=(__bf16)f3.w;
    ax[kt] = a;
  }

  // ---- Gates: logits[16x8] = x @ [g1_W | g2_W]  (2 interleaved acc chains) ----
  v8f gl0 = zacc, gl1 = zacc;
  #pragma unroll
  for (int kt = 0; kt < 16; kt += 2) {
    gl0 = wmma_bf16(ax[kt],     bfrag(P.blob + OFF_G, kt,     lane), gl0);
    gl1 = wmma_bf16(ax[kt + 1], bfrag(P.blob + OFF_G, kt + 1, lane), gl1);
  }
  float gbias = 0.f;
  if (col < 4)      gbias = P.g1b[col];
  else if (col < 8) gbias = P.g2b[col - 4];
  float sel[8];
  #pragma unroll
  for (int r = 0; r < 8; ++r) {
    float l = gl0[r] + gl1[r] + gbias;
    float m = fmaxf(l, __shfl_xor(l, 1));
    m = fmaxf(m, __shfl_xor(m, 2));
    float e = __expf(l - m);
    float s = e + __shfl_xor(e, 1);
    s += __shfl_xor(s, 2);
    sel[r] = e / s;
  }

  v8f g1acc[2] = {zacc, zacc};
  v8f g2acc[2] = {zacc, zacc};

  // ---- 6 experts: 0,1 shared | 2,3 task1 | 4,5 task2 ----
  for (int ex = 0; ex < 6; ++ex) {
    const int grp = ex >> 1, sub = ex & 1;
    const float* b1 = (grp == 0 ? P.sh_b1 : grp == 1 ? P.t1_b1 : P.t2_b1) + sub * 64;
    const float* b2 = (grp == 0 ? P.sh_b2 : grp == 1 ? P.t1_b2 : P.t2_b2) + sub * 32;
    const float* b3 = (grp == 0 ? P.sh_b3 : grp == 1 ? P.t1_b3 : P.t2_b3) + sub * 32;
    const __bf16* w1 = P.blob + OFF_W1 + ex * 32768;
    const __bf16* w2 = P.blob + OFF_W2 + ex * 2048;
    const __bf16* w3 = P.blob + OFF_W3 + ex * 1024;

    // Layer 1: h1 = relu(x @ W1 + b1)  [16x64] -> LDS
    // kt-outer / nt-inner: 4 independent accumulator chains, A reused 4x
    {
      v8f acc[4] = {zacc, zacc, zacc, zacc};
      #pragma unroll
      for (int kt = 0; kt < 16; ++kt) {
        #pragma unroll
        for (int nt = 0; nt < 4; ++nt)
          acc[nt] = wmma_bf16(ax[kt], bfrag(w1, kt * 4 + nt, lane), acc[nt]);
      }
      #pragma unroll
      for (int nt = 0; nt < 4; ++nt) {
        float bb = b1[nt * 16 + col];
        #pragma unroll
        for (int r = 0; r < 8; ++r)
          hw[(r + 8 * half) * LDH1 + nt * 16 + col] =
              (__bf16)fmaxf(acc[nt][r] + bb, 0.f);
      }
    }
    // Layer 2: h2 = relu(h1 @ W2 + b2)  [16x32] -> LDS (2 independent chains)
    {
      v8f acc[2] = {zacc, zacc};
      #pragma unroll
      for (int kt = 0; kt < 2; ++kt) {
        v16bf a = lds_afrag(hw + col * LDH1, kt * 32 + khalf);
        #pragma unroll
        for (int nt = 0; nt < 2; ++nt)
          acc[nt] = wmma_bf16(a, bfrag(w2, kt * 2 + nt, lane), acc[nt]);
      }
      #pragma unroll
      for (int nt = 0; nt < 2; ++nt) {
        float bb = b2[nt * 16 + col];
        #pragma unroll
        for (int r = 0; r < 8; ++r)
          h2w[(r + 8 * half) * LDH2 + nt * 16 + col] =
              (__bf16)fmaxf(acc[nt][r] + bb, 0.f);
      }
    }
    // Layer 3: eo = h2 @ W3 + b3, then gate-weighted accumulation
    const int g1col = (ex < 2) ? (2 + ex) : (ex < 4 ? ex - 2 : -1);
    const int g2col = (ex < 2) ? (6 + ex) : (ex >= 4 ? ex : -1);  // ex-4+4
    {
      v16bf a = lds_afrag(h2w + col * LDH2, khalf);
      #pragma unroll
      for (int nt = 0; nt < 2; ++nt) {
        v8f acc = wmma_bf16(a, bfrag(w3, nt, lane), zacc);
        float bb = b3[nt * 16 + col];
        #pragma unroll
        for (int r = 0; r < 8; ++r) {
          float v = acc[r] + bb;
          if (g1col >= 0) g1acc[nt][r] += __shfl(sel[r], (lane & 16) | g1col) * v;
          if (g2col >= 0) g2acc[nt][r] += __shfl(sel[r], (lane & 16) | g2col) * v;
        }
      }
    }
  }

  // ---- Task-1 tower: relu(g1 @ W1 + b1) @ W2 + b2 ----
  #pragma unroll
  for (int nt = 0; nt < 2; ++nt)
    #pragma unroll
    for (int r = 0; r < 8; ++r)
      h2w[(r + 8 * half) * LDH2 + nt * 16 + col] = (__bf16)g1acc[nt][r];
  {
    v16bf a = lds_afrag(h2w + col * LDH2, khalf);
    #pragma unroll
    for (int nt = 0; nt < 4; ++nt) {
      v8f acc = wmma_bf16(a, bfrag(P.blob + OFF_TW1A, nt, lane), zacc);
      float bb = P.tw1b1[nt * 16 + col];
      #pragma unroll
      for (int r = 0; r < 8; ++r)
        hw[(r + 8 * half) * LDH1 + nt * 16 + col] =
            (__bf16)fmaxf(acc[r] + bb, 0.f);
    }
  }
  v8f o1 = zacc;
  #pragma unroll
  for (int kt = 0; kt < 2; ++kt)
    o1 = wmma_bf16(lds_afrag(hw + col * LDH1, kt * 32 + khalf),
                   bfrag(P.blob + OFF_TW1B, kt, lane), o1);

  // ---- Task-2 tower (all WMMAs before any divergent stores) ----
  #pragma unroll
  for (int nt = 0; nt < 2; ++nt)
    #pragma unroll
    for (int r = 0; r < 8; ++r)
      h2w[(r + 8 * half) * LDH2 + nt * 16 + col] = (__bf16)g2acc[nt][r];
  {
    v16bf a = lds_afrag(h2w + col * LDH2, khalf);
    #pragma unroll
    for (int nt = 0; nt < 4; ++nt) {
      v8f acc = wmma_bf16(a, bfrag(P.blob + OFF_TW2A, nt, lane), zacc);
      float bb = P.tw2b1[nt * 16 + col];
      #pragma unroll
      for (int r = 0; r < 8; ++r)
        hw[(r + 8 * half) * LDH1 + nt * 16 + col] =
            (__bf16)fmaxf(acc[r] + bb, 0.f);
    }
  }
  v8f o2 = zacc;
  #pragma unroll
  for (int kt = 0; kt < 2; ++kt)
    o2 = wmma_bf16(lds_afrag(hw + col * LDH1, kt * 32 + khalf),
                   bfrag(P.blob + OFF_TW2B, kt, lane), o2);

  // ---- Outputs ----
  if (col < 3) {
    float bb = P.tw1b2[col];
    #pragma unroll
    for (int r = 0; r < 8; ++r) {
      float v = o1[r] + bb;
      if (col != 1) v = 1.f / (1.f + __expf(-v));   // sigmoid cols 0 and 2
      P.out[(size_t)(rowBase + r + 8 * half) * 3 + col] = v;
    }
  }
  if (col < 2) {
    float bb = P.tw2b2[col];
    float* out2 = P.out + (size_t)B_ROWS * 3;
    #pragma unroll
    for (int r = 0; r < 8; ++r) {
      float v = o2[r] + bb;
      if (col == 1) v = 1.f / (1.f + __expf(-v));   // sigmoid col 1
      out2[(size_t)(rowBase + r + 8 * half) * 2 + col] = v;
    }
  }
}

extern "C" void kernel_launch(void* const* d_in, const int* in_sizes, int n_in,
                              void* d_out, int out_size, void* d_ws, size_t ws_size,
                              hipStream_t stream) {
  (void)in_sizes; (void)n_in; (void)out_size; (void)ws_size;
  __bf16* blob = (__bf16*)d_ws;
  const float* in[31];
  for (int i = 0; i < 31; ++i) in[i] = (const float*)d_in[i];

  // Pack all weights into WMMA-B-fragment bf16 blobs (weights are tiny, L2-hot).
  for (int g = 0; g < 3; ++g) {         // 0=shared, 1=task1, 2=task2
    const float* W1 = in[1 + 6 * g];
    const float* W2 = in[3 + 6 * g];
    const float* W3 = in[5 + 6 * g];
    for (int e = 0; e < 2; ++e) {
      int ex = g * 2 + e;
      pack_b<<<128, 256, 0, stream>>>(W1 + e * 512 * 64, blob + OFF_W1 + ex * 32768, 512, 64, 4);
      pack_b<<<8,   256, 0, stream>>>(W2 + e * 64 * 32,  blob + OFF_W2 + ex * 2048,  64,  32, 2);
      pack_b<<<4,   256, 0, stream>>>(W3 + e * 32 * 32,  blob + OFF_W3 + ex * 1024,  32,  32, 2);
    }
  }
  pack_gate<<<32, 256, 0, stream>>>(in[19], in[21], blob + OFF_G);
  pack_b<<<8, 256, 0, stream>>>(in[23], blob + OFF_TW1A, 32, 64, 4);
  pack_b<<<8, 256, 0, stream>>>(in[27], blob + OFF_TW2A, 32, 64, 4);
  pack_b<<<4, 256, 0, stream>>>(in[25], blob + OFF_TW1B, 64, 3, 1);
  pack_b<<<4, 256, 0, stream>>>(in[29], blob + OFF_TW2B, 64, 2, 1);

  PLEParams P;
  P.x = in[0]; P.blob = blob;
  P.sh_b1 = in[2];  P.t1_b1 = in[8];  P.t2_b1 = in[14];
  P.sh_b2 = in[4];  P.t1_b2 = in[10]; P.t2_b2 = in[16];
  P.sh_b3 = in[6];  P.t1_b3 = in[12]; P.t2_b3 = in[18];
  P.g1b = in[20];   P.g2b = in[22];
  P.tw1b1 = in[24]; P.tw1b2 = in[26];
  P.tw2b1 = in[28]; P.tw2b2 = in[30];
  P.out = (float*)d_out;

  ple_fused<<<B_ROWS / (16 * WAVES), 32 * WAVES, 0, stream>>>(P);
}
